// Net_81080392614027
// MI455X (gfx1250) — matrix operand order repaired and checked by reading, not compiled, not run on hardware
//
#include <hip/hip_runtime.h>
#include <hip/hip_bf16.h>
#include <math.h>

// ---------------- problem constants ----------------
constexpr int NN  = 10000;  // nodes
constexpr int NE  = 80000;  // edges  (80000 % 64 == 0 -> no WMMA tail)
constexpr int HID = 25;     // edge-MLP hidden width
constexpr int HS  = 26;     // h row stride: 25 hidden + baked-in 1.0 bias column

// ---------------- CDNA5 WMMA / TDM types ----------------
typedef __attribute__((ext_vector_type(16))) __bf16 v16bf;
typedef __attribute__((ext_vector_type(8)))  float  v8f;
typedef unsigned int u32x4 __attribute__((ext_vector_type(4)));
typedef int          i32x4 __attribute__((ext_vector_type(4)));
typedef int          i32x8 __attribute__((ext_vector_type(8)));

// relaxed agent-scope float atomic add (lowers to global_atomic_add_f32)
__device__ __forceinline__ void atomAddF(float* p, float v) {
    __hip_atomic_fetch_add(p, v, __ATOMIC_RELAXED, __HIP_MEMORY_SCOPE_AGENT);
}

__device__ __forceinline__ float elu1(float x) {
    return x > 0.f ? x : (expf(x) - 1.f);
}

// Issue a TDM 2D load of `nElem4B` consecutive 4-byte elements from `gaddr`
// into LDS at byte offset `ldsAddr` (D# layout per CDNA5 ISA §8.3/8.4).
__device__ __forceinline__ void tdm_load_1d(unsigned ldsAddr,
                                            unsigned long long gaddr,
                                            unsigned nElem4B) {
    u32x4 g0;
    g0[0] = 1u;                                   // count=1, no gather
    g0[1] = ldsAddr;                              // lds_addr (bytes)
    g0[2] = (unsigned)gaddr;                      // global_addr[31:0]
    g0[3] = (unsigned)((gaddr >> 32) & 0x01FFFFFFu) | (2u << 30); // addr[56:32] | type=2
    i32x8 g1;
    g1[0] = (int)(2u << 16);                      // workgroup_mask=0, data_size=4B
    g1[1] = (int)((nElem4B & 0xFFFFu) << 16);     // tensor_dim0[15:0]
    g1[2] = (int)((nElem4B >> 16) | (1u << 16));  // tensor_dim0[31:16] | tensor_dim1=1
    g1[3] = (int)(nElem4B << 16);                 // tile_dim0 (fits 16 bits)
    g1[4] = 1;                                    // tile_dim1=1, tile_dim2=0
    g1[5] = (int)nElem4B;                         // tensor_dim0_stride[31:0]
    g1[6] = 0;
    g1[7] = 0;
    i32x4 z4 = {0, 0, 0, 0};
#if __clang_major__ >= 23
    i32x8 z8 = {0, 0, 0, 0, 0, 0, 0, 0};
    __builtin_amdgcn_tensor_load_to_lds(g0, g1, z4, z4, z8, 0);
#else
    __builtin_amdgcn_tensor_load_to_lds(g0, g1, z4, z4, 0);
#endif
}

// ---------------- workspace layout (floats) ----------------
constexpr size_t OFF_H1   = 0;                            // NE*26
constexpr size_t OFF_H2   = OFF_H1   + (size_t)NE * HS;
constexpr size_t OFF_AGG1 = OFF_H2   + (size_t)NE * HS;   // NN*32
constexpr size_t OFF_AGG2 = OFF_AGG1 + (size_t)NN * 32;   // NN*64
constexpr size_t OFF_DEG  = OFF_AGG2 + (size_t)NN * 64;   // NN
constexpr size_t OFF_Y1   = OFF_DEG  + (size_t)NN;        // NN*32
constexpr size_t OFF_Y2   = OFF_Y1   + (size_t)NN * 32;   // NN*64
constexpr size_t OFF_X2   = OFF_Y2   + (size_t)NN * 64;   // NN*32
constexpr size_t OFF_X3   = OFF_X2   + (size_t)NN * 32;   // NN*64
constexpr size_t OFF_BN1S = OFF_X3   + (size_t)NN * 64;   // 32+32+64+64
constexpr size_t OFF_BN2S = OFF_BN1S + 64;
constexpr size_t FLT_END  = OFF_BN2S + 128;
constexpr size_t BPK1_BYTE = ((FLT_END * 4 + 255) / 256) * 256; // bf16 [416*32]
constexpr size_t BPK2_BYTE = BPK1_BYTE + (size_t)416 * 32 * 2;  // bf16 [832*64]

// ============================================================
// K0: pack Wcat = [W2 reshaped ; B2] into WMMA B-fragment order.
// B (KxN=32x16, bf16): lanes 0-15 -> N, K=j (j<8), 16+(j-8) (j>=8) ; lanes 16-31 -> K+8.
// Storage: [(ntile*KB + kb32)*32 + lane]*16 + j  (32B contiguous per lane)
// ============================================================
template<int CIN, int COUT, int KTOT>
__global__ void k_pack_b(const float* __restrict__ w2,
                         const float* __restrict__ b2,
                         __bf16* __restrict__ bpk) {
    constexpr int KB = KTOT / 32;
    int t = blockIdx.x * blockDim.x + threadIdx.x;
    if (t >= KTOT * COUT) return;
    int j    = t & 15;
    int lane = (t >> 4) & 31;
    int rest = t >> 9;
    int kb32 = rest % KB;
    int nt   = rest / KB;
    int v    = j >> 1, e = j & 1;
    int hi   = lane >> 4;
    int k    = kb32 * 32 + (hi ? 8 : 0) + ((v < 4) ? (2 * v + e) : (2 * v + e + 8)); // row of Wcat
    int c    = nt * 16 + (lane & 15);                 // col of Wcat
    float val;
    if (k < HID * CIN) {
        int khid = k / CIN, i = k % CIN;
        val = w2[(size_t)khid * (CIN * COUT) + i * COUT + c];
    } else {
        val = b2[(k - HID * CIN) * COUT + c];
    }
    bpk[t] = (__bf16)val;
}

// ============================================================
// K1: edge MLP layer-1 for both convs (relu(ea@w1+b1)) + degree count.
// Row stride 26; column 25 = 1.0 (bias rows of the outer-product GEMM).
// ============================================================
__global__ void k_edge_mlp(const float* __restrict__ ea,
                           const int*   __restrict__ dst,
                           const float* __restrict__ w1a, const float* __restrict__ b1a,
                           const float* __restrict__ w1b, const float* __restrict__ b1b,
                           float* __restrict__ h1, float* __restrict__ h2,
                           float* __restrict__ deg) {
    int e = blockIdx.x * blockDim.x + threadIdx.x;
    if (e >= NE) return;
    float a[8];
#pragma unroll
    for (int j = 0; j < 8; ++j) a[j] = ea[e * 8 + j];
#pragma unroll
    for (int k = 0; k < HID; ++k) {
        float s1 = b1a[k], s2 = b1b[k];
#pragma unroll
        for (int j = 0; j < 8; ++j) {
            s1 = fmaf(a[j], w1a[j * HID + k], s1);
            s2 = fmaf(a[j], w1b[j * HID + k], s2);
        }
        h1[(size_t)e * HS + k] = fmaxf(s1, 0.f);
        h2[(size_t)e * HS + k] = fmaxf(s2, 0.f);
    }
    h1[(size_t)e * HS + HID] = 1.0f;
    h2[(size_t)e * HS + HID] = 1.0f;
    atomAddF(&deg[dst[e]], 1.0f);
}

// ============================================================
// K2: fused NNConv message GEMM.
//   msg = (h ⊗ x_src | x_src) @ Wcat   via v_wmma_f32_16x16x32_bf16.
// h-tile (contiguous 64*26 floats) DMA'd into LDS by the Tensor Data Mover;
// A fragments synthesized in-register; C scattered with f32 atomics.
// 4 waves/block, one 16-edge M-tile per wave, NT N-tiles, K-loop of KTOT/32.
// ============================================================
template<int CIN, int COUT, int KTOT, int L2CIN>
__global__ __launch_bounds__(128)
void k_msg_wmma(const float* __restrict__ xin,
                const float* __restrict__ h,
                const int*   __restrict__ src,
                const int*   __restrict__ dst,
                const __bf16* __restrict__ bpk,
                float* __restrict__ agg) {
    constexpr int KB = KTOT / 32;
    constexpr int NT = COUT / 16;
    __shared__ float xs[4][16][CIN];
    __shared__ float hblock[64 * HS];     // block's 64 edge-hidden rows (via TDM)

    const int w     = threadIdx.x >> 5;
    const int lane  = threadIdx.x & 31;
    const int ebase = blockIdx.x * 64 + w * 16;

    // wave 0: one TDM descriptor pulls all 64 h-rows (6656 B) into LDS
    if (threadIdx.x < 32) {
        tdm_load_1d((unsigned)(size_t)&hblock[0],
                    (unsigned long long)(size_t)(h + (size_t)blockIdx.x * 64 * HS),
                    64u * HS);
        __builtin_amdgcn_s_wait_tensorcnt(0);
    }
    // gather x[src] rows (indexed -> regular vector loads)
    if (lane < 16) {
        int s = src[ebase + lane];
        for (int j = 0; j < CIN; ++j) xs[w][lane][j] = xin[(size_t)s * CIN + j];
    }
    __builtin_prefetch(bpk, 0, 3);        // global_prefetch_b8 warm-up
    __syncthreads();

    v8f acc[NT];
#pragma unroll
    for (int nt = 0; nt < NT; ++nt) {
        v8f z = {0.f, 0.f, 0.f, 0.f, 0.f, 0.f, 0.f, 0.f};
        acc[nt] = z;
    }

    const int m       = lane & 15;
    const int hi      = lane >> 4;
    const int rowbase = (w * 16 + m) * HS;
    for (int kb32 = 0; kb32 < KB; ++kb32) {
        const int kb = kb32 * 32;
        // --- synthesize A fragment (16-bit A 16x32 layout, ISA §7.12.2) ---
        v16bf a;
#pragma unroll
        for (int v = 0; v < 8; ++v) {
            int klo = ((v < 4) ? (2 * v) : (2 * v + 8)) + (hi ? 8 : 0);
            int k0  = kb + klo;
            float f0 = hblock[rowbase + (k0 >> L2CIN)]       * xs[w][m][k0 & (CIN - 1)];
            float f1 = hblock[rowbase + ((k0 + 1) >> L2CIN)] * xs[w][m][(k0 + 1) & (CIN - 1)];
            a[2 * v]     = (__bf16)f0;
            a[2 * v + 1] = (__bf16)f1;
        }
#pragma unroll
        for (int nt = 0; nt < NT; ++nt) {
            v16bf b = *(const v16bf*)(bpk + ((size_t)(nt * KB + kb32) * 32 + lane) * 16);
            acc[nt] = __builtin_amdgcn_wmma_f32_16x16x32_bf16(
                false, a, false, b, (short)0, acc[nt], false, false);
        }
    }

    // --- scatter C (f32 C/D layout: VGPR v -> M=v / v+8) into agg[dst] ---
    // 32-bit unsigned offsets -> saddr-base + voffset atomic addressing
#pragma unroll
    for (int nt = 0; nt < NT; ++nt) {
#pragma unroll
        for (int v = 0; v < 8; ++v) {
            int mrow = v + (hi ? 8 : 0);
            unsigned node = (unsigned)dst[ebase + mrow];
            unsigned off  = node * COUT + nt * 16 + (lane & 15);
            atomAddF(agg + off, acc[nt][v]);
        }
    }
}

// ============================================================
// K3: node update: elu(x@root + agg/deg + bias), accumulate BN stats
// ============================================================
template<int CIN, int COUT>
__global__ void k_node_update(const float* __restrict__ xin,
                              const float* __restrict__ root,
                              const float* __restrict__ bias,
                              const float* __restrict__ agg,
                              const float* __restrict__ deg,
                              float* __restrict__ y,
                              float* __restrict__ bnsum,
                              float* __restrict__ bnsq) {
    int n = blockIdx.x;
    int o = threadIdx.x;           // blockDim == COUT
    float s = bias[o];
#pragma unroll
    for (int i = 0; i < CIN; ++i)
        s = fmaf(xin[n * CIN + i], root[i * COUT + o], s);
    float d = deg[n];
    d = d < 1.f ? 1.f : d;
    s += agg[n * COUT + o] / d;
    float v = elu1(s);
    y[n * COUT + o] = v;
    atomAddF(&bnsum[o], v);
    atomAddF(&bnsq[o], v * v);
}

// ============================================================
// K4: BN normalize (training stats) + ELU
// ============================================================
template<int COUT>
__global__ void k_bn_elu(const float* __restrict__ y,
                         const float* __restrict__ bnsum,
                         const float* __restrict__ bnsq,
                         const float* __restrict__ gamma,
                         const float* __restrict__ beta,
                         float* __restrict__ xo) {
    int t = blockIdx.x * blockDim.x + threadIdx.x;
    if (t >= NN * COUT) return;
    int o = t % COUT;
    float m   = bnsum[o] * (1.0f / NN);
    float var = bnsq[o]  * (1.0f / NN) - m * m;
    float val = (y[t] - m) * rsqrtf(var + 1e-5f) * gamma[o] + beta[o];
    xo[t] = elu1(val);
}

// ============================================================
// K5: per-node FC head (64->128->1, ELU) + global sum reduction
// ============================================================
__global__ __launch_bounds__(128)
void k_fc_reduce(const float* __restrict__ x3,
                 const float* __restrict__ w1, const float* __restrict__ b1,
                 const float* __restrict__ w2, const float* __restrict__ b2,
                 float* __restrict__ out) {
    __shared__ float row[64];
    __shared__ float part[128];
    int n = blockIdx.x;
    int t = threadIdx.x;
    if (t < 64) row[t] = x3[n * 64 + t];
    __syncthreads();
    float s = b1[t];
#pragma unroll
    for (int i = 0; i < 64; ++i) s = fmaf(row[i], w1[i * 128 + t], s);
    s = elu1(s);
    part[t] = s * w2[t];
    __syncthreads();
    for (int off = 64; off > 0; off >>= 1) {
        if (t < off) part[t] += part[t + off];
        __syncthreads();
    }
    if (t == 0) {
        float r = elu1(part[0] + b2[0]);
        atomAddF(out, r);
    }
}

// ============================================================
extern "C" void kernel_launch(void* const* d_in, const int* in_sizes, int n_in,
                              void* d_out, int out_size, void* d_ws, size_t ws_size,
                              hipStream_t stream) {
    const float* x          = (const float*)d_in[0];
    const int*   eidx       = (const int*)  d_in[1];
    const float* edge_attr  = (const float*)d_in[2];
    const float* nn1_w1     = (const float*)d_in[3];
    const float* nn1_b1     = (const float*)d_in[4];
    const float* nn1_w2     = (const float*)d_in[5];
    const float* nn1_b2     = (const float*)d_in[6];
    const float* conv1_root = (const float*)d_in[7];
    const float* conv1_bias = (const float*)d_in[8];
    const float* bn1_gamma  = (const float*)d_in[9];
    const float* bn1_beta   = (const float*)d_in[10];
    const float* nn2_w1     = (const float*)d_in[11];
    const float* nn2_b1     = (const float*)d_in[12];
    const float* nn2_w2     = (const float*)d_in[13];
    const float* nn2_b2     = (const float*)d_in[14];
    const float* conv2_root = (const float*)d_in[15];
    const float* conv2_bias = (const float*)d_in[16];
    const float* bn2_gamma  = (const float*)d_in[17];
    const float* bn2_beta   = (const float*)d_in[18];
    const float* fc1_w      = (const float*)d_in[19];
    const float* fc1_b      = (const float*)d_in[20];
    const float* fc2_w      = (const float*)d_in[21];
    const float* fc2_b      = (const float*)d_in[22];

    const int* src = eidx;
    const int* dst = eidx + NE;

    float* ws   = (float*)d_ws;
    float* h1   = ws + OFF_H1;
    float* h2   = ws + OFF_H2;
    float* agg1 = ws + OFF_AGG1;
    float* agg2 = ws + OFF_AGG2;
    float* deg  = ws + OFF_DEG;
    float* y1   = ws + OFF_Y1;
    float* y2   = ws + OFF_Y2;
    float* x2   = ws + OFF_X2;
    float* x3   = ws + OFF_X3;
    float* bn1s = ws + OFF_BN1S;   // bn1 sum/sq + bn2 sum/sq contiguous (192 floats)
    float* bn2s = ws + OFF_BN2S;
    __bf16* bpk1 = (__bf16*)((char*)d_ws + BPK1_BYTE);
    __bf16* bpk2 = (__bf16*)((char*)d_ws + BPK2_BYTE);

    // zero accumulators (graph-capture-safe async memsets)
    hipMemsetAsync(agg1, 0, (size_t)NN * 32 * sizeof(float), stream);
    hipMemsetAsync(agg2, 0, (size_t)NN * 64 * sizeof(float), stream);
    hipMemsetAsync(deg,  0, (size_t)NN * sizeof(float), stream);
    hipMemsetAsync(bn1s, 0, 192 * sizeof(float), stream);
    hipMemsetAsync(d_out, 0, sizeof(float), stream);

    // pack B operands into WMMA fragment order (bf16)
    k_pack_b<16, 32, 416><<<(416 * 32 + 255) / 256, 256, 0, stream>>>(nn1_w2, nn1_b2, bpk1);
    k_pack_b<32, 64, 832><<<(832 * 64 + 255) / 256, 256, 0, stream>>>(nn2_w2, nn2_b2, bpk2);

    // edge MLP layer-1 (both convs) + degree
    k_edge_mlp<<<(NE + 255) / 256, 256, 0, stream>>>(
        edge_attr, dst, nn1_w1, nn1_b1, nn2_w1, nn2_b1, h1, h2, deg);

    // ---- conv1 ----
    k_msg_wmma<16, 32, 416, 4><<<NE / 64, 128, 0, stream>>>(x, h1, src, dst, bpk1, agg1);
    k_node_update<16, 32><<<NN, 32, 0, stream>>>(x, conv1_root, conv1_bias, agg1, deg,
                                                 y1, bn1s, bn1s + 32);
    k_bn_elu<32><<<(NN * 32 + 255) / 256, 256, 0, stream>>>(y1, bn1s, bn1s + 32,
                                                            bn1_gamma, bn1_beta, x2);
    // ---- conv2 ----
    k_msg_wmma<32, 64, 832, 5><<<NE / 64, 128, 0, stream>>>(x2, h2, src, dst, bpk2, agg2);
    k_node_update<32, 64><<<NN, 64, 0, stream>>>(x2, conv2_root, conv2_bias, agg2, deg,
                                                 y2, bn2s, bn2s + 64);
    k_bn_elu<64><<<(NN * 64 + 255) / 256, 256, 0, stream>>>(y2, bn2s, bn2s + 64,
                                                            bn2_gamma, bn2_beta, x3);
    // ---- FC head + graph readout ----
    k_fc_reduce<<<NN, 128, 0, stream>>>(x3, fc1_w, fc1_b, fc2_w, fc2_b, (float*)d_out);
}
